// Model_20401094656117
// MI455X (gfx1250) — matrix operand order repaired and checked by reading, not compiled, not run on hardware
//
#include <hip/hip_runtime.h>
#include <hip/hip_bf16.h>

typedef __attribute__((ext_vector_type(16))) __bf16 v16bf;
typedef __attribute__((ext_vector_type(8)))  float  v8f;

#define DIM 64
#define SD  192  // (L+1)*D, L=2

// ---- degree count: deg[dst[e]] += 1 ----
__global__ void k_deg(const int* __restrict__ dst, float* __restrict__ deg, int E) {
  int e = blockIdx.x * blockDim.x + threadIdx.x;
  if (e < E) unsafeAtomicAdd(&deg[dst[e]], 1.0f);
}

// ---- deg -> dinv = deg>0 ? rsqrt(max(deg,1)) : 0 (in place) ----
__global__ void k_rsqrt(float* __restrict__ d, int n) {
  int i = blockIdx.x * blockDim.x + threadIdx.x;
  if (i < n) {
    float v = d[i];
    d[i] = (v > 0.f) ? rsqrtf(fmaxf(v, 1.f)) : 0.f;
  }
}

// ---- init s_all: block0 = [user_emb; src_item_emb], blocks 1,2 = 0 ----
__global__ void k_init_s(const float* __restrict__ ue, const float* __restrict__ ie,
                         float* __restrict__ s_all, int NU, int ns) {
  int tid = blockIdx.x * blockDim.x + threadIdx.x;  // ns*48 threads (48 float4/row)
  int row = tid / 48, c = tid % 48;
  if (row >= ns) return;
  float4 v = make_float4(0.f, 0.f, 0.f, 0.f);
  if (c < 16) {
    const float* srcr = (row < NU) ? (ue + (size_t)row * DIM)
                                   : (ie + (size_t)(row - NU) * DIM);
    v = *(const float4*)(srcr + c * 4);
  }
  *(float4*)(s_all + (size_t)row * SD + c * 4) = v;
}

// ---- init tx_a = [user_emb; tgt_item_emb] ----
__global__ void k_init_t(const float* __restrict__ ue, const float* __restrict__ ie,
                         float* __restrict__ tx, int NU, int nt) {
  int tid = blockIdx.x * blockDim.x + threadIdx.x;  // nt*16 threads
  int row = tid / 16, c = tid % 16;
  if (row >= nt) return;
  const float* srcr = (row < NU) ? (ue + (size_t)row * DIM)
                                 : (ie + (size_t)(row - NU) * DIM);
  *(float4*)(tx + (size_t)row * DIM + c * 4) =
      *(const float4*)(srcr + c * 4);
}

// ---- GCN conv: out[dst] += in[src] * dinv[src]*dinv[dst]; 16 threads/edge ----
__global__ void k_conv(const float* __restrict__ xin, float* __restrict__ xout,
                       const int* __restrict__ src, const int* __restrict__ dst,
                       const float* __restrict__ dinv, int E,
                       int stride_in, int stride_out) {
  unsigned long long gid = (unsigned long long)blockIdx.x * blockDim.x + threadIdx.x;
  int e = (int)(gid >> 4);
  if (e >= E) return;
  int c = (int)(gid & 15);
  int s = src[e], d = dst[e];
  float w = dinv[s] * dinv[d];
  float4 v = *(const float4*)(xin + (size_t)s * stride_in + c * 4);
  float* o = xout + (size_t)d * stride_out + c * 4;
  unsafeAtomicAdd(o + 0, v.x * w);
  unsafeAtomicAdd(o + 1, v.y * w);
  unsafeAtomicAdd(o + 2, v.z * w);
  unsafeAtomicAdd(o + 3, v.w * w);
}

// ---- mix: u = [s_blk[:NU] | t_blk[:NU]] @ W^T + b ; write u back into both.
// One wave computes a 16-row x 64-col tile with bf16 WMMA (f32 accumulate).
__global__ void k_mix(const float* __restrict__ W,     // [64,128] (row n = out dim)
                      const float* __restrict__ bias,  // [64]
                      float* __restrict__ s_blk,       // stride SD, cols = first 64
                      float* __restrict__ t_blk,       // stride DIM
                      int NU) {
  int wave = (blockIdx.x * blockDim.x + threadIdx.x) >> 5;
  int lane = threadIdx.x & 31;
  int tiles = (NU + 15) / 16;
  if (wave >= tiles) return;                // wave-uniform exit; EXEC stays full
  int row0 = wave * 16;
  int half = lane >> 4;                     // 0: lanes 0-15, 1: lanes 16-31
  int lr = lane & 15;
  int rowA = row0 + lr;
  if (rowA >= NU) rowA = NU - 1;            // clamp (NU%16==0 normally)

  const float* srow = s_blk + (size_t)rowA * SD;
  const float* trow = t_blk + (size_t)rowA * DIM;

  v8f c0 = {}, c1 = {}, c2 = {}, c3 = {};
#pragma unroll
  for (int ks = 0; ks < 4; ++ks) {          // K = 128 in 4 steps of 32
    v16bf a;
#pragma unroll
    for (int j = 0; j < 8; ++j) {
      int k0 = ks * 32 + half * 8 + j;
      int k1 = ks * 32 + 16 + half * 8 + j;
      float f0 = (k0 < DIM) ? srow[k0] : trow[k0 - DIM];
      float f1 = (k1 < DIM) ? srow[k1] : trow[k1 - DIM];
      a[j]     = (__bf16)f0;
      a[8 + j] = (__bf16)f1;
    }
    v8f* cc[4] = {&c0, &c1, &c2, &c3};
#pragma unroll
    for (int nt = 0; nt < 4; ++nt) {
      v16bf b;
      int n = nt * 16 + lr;
      const float* wrow = W + (size_t)n * 128 + ks * 32 + half * 8;
#pragma unroll
      for (int j = 0; j < 8; ++j) { b[j] = (__bf16)wrow[j]; b[8 + j] = (__bf16)wrow[16 + j]; }
      *cc[nt] = __builtin_amdgcn_wmma_f32_16x16x32_bf16(
          false, a, false, b, (short)0, *cc[nt], false, false);
    }
  }

  // C/D layout: VGPR r -> M = r (lanes 0-15) or 8+r (lanes 16-31), N = lane%16
  v8f cs[4] = {c0, c1, c2, c3};
#pragma unroll
  for (int nt = 0; nt < 4; ++nt) {
#pragma unroll
    for (int r = 0; r < 8; ++r) {
      int m = row0 + r + half * 8;
      int n = nt * 16 + lr;
      if (m < NU) {
        float v = cs[nt][r] + bias[n];
        s_blk[(size_t)m * SD + n] = v;
        t_blk[(size_t)m * DIM + n] = v;
      }
    }
  }
}

// ---- prediction: one wave per link ----
__global__ void k_pred(const float* __restrict__ s_all,
                       const int* __restrict__ lu, const int* __restrict__ li,
                       const float* __restrict__ pw, const float* __restrict__ pb,
                       float* __restrict__ out, int B) {
  int wave = (blockIdx.x * blockDim.x + threadIdx.x) >> 5;
  int lane = threadIdx.x & 31;
  if (wave >= B) return;
  const float* ur = s_all + (size_t)lu[wave] * SD;
  const float* ir = s_all + (size_t)li[wave] * SD;
  float acc = 0.f;
#pragma unroll
  for (int k = lane; k < SD; k += 32)
    acc += pw[k] * ur[k] + pw[SD + k] * ir[k];
#pragma unroll
  for (int off = 16; off; off >>= 1) acc += __shfl_xor(acc, off, 32);
  if (lane == 0) {
    float x = acc + pb[0];
    x = (x >= 0.f) ? x : 0.01f * x;          // leaky_relu(0.01)
    out[wave] = 1.f / (1.f + expf(-x));      // sigmoid
  }
}

extern "C" void kernel_launch(void* const* d_in, const int* in_sizes, int n_in,
                              void* d_out, int out_size, void* d_ws, size_t ws_size,
                              hipStream_t stream) {
  const int*   se     = (const int*)d_in[0];
  const int*   te     = (const int*)d_in[1];
  const int*   link   = (const int*)d_in[2];
  const float* ue     = (const float*)d_in[3];
  const float* sie    = (const float*)d_in[4];
  const float* tie    = (const float*)d_in[5];
  const float* mix_W  = (const float*)d_in[6];
  const float* mix_b  = (const float*)d_in[7];
  const float* pred_W = (const float*)d_in[8];
  const float* pred_b = (const float*)d_in[9];

  const int NU  = in_sizes[3] / DIM;
  const int NSI = in_sizes[4] / DIM;
  const int NTI = in_sizes[5] / DIM;
  const int ns = NU + NSI, nt = NU + NTI;
  const int E  = in_sizes[0] / 2;
  const int Et = in_sizes[1] / 2;
  const int B  = in_sizes[2] / 2;

  const int* s_src = se;       const int* s_dst = se + E;
  const int* t_src = te;       const int* t_dst = te + Et;
  const int* l_u   = link;     const int* l_i   = link + B;

  float* ws = (float*)d_ws;
  float* s_all  = ws;                    size_t off = (size_t)ns * SD;
  float* tx_a   = ws + off;              off += (size_t)nt * DIM;
  float* tx_b   = ws + off;              off += (size_t)nt * DIM;
  float* dinv_s = ws + off;              off += (size_t)ns;
  float* dinv_t = ws + off;              off += (size_t)nt;

  const int T = 256;
  // --- degrees ---
  hipMemsetAsync(dinv_s, 0, (size_t)(ns + nt) * sizeof(float), stream);
  k_deg<<<(E + T - 1) / T, T, 0, stream>>>(s_dst, dinv_s, E);
  k_deg<<<(Et + T - 1) / T, T, 0, stream>>>(t_dst, dinv_t, Et);
  k_rsqrt<<<(ns + T - 1) / T, T, 0, stream>>>(dinv_s, ns);
  k_rsqrt<<<(nt + T - 1) / T, T, 0, stream>>>(dinv_t, nt);

  // --- init features (also zeroes s_all blocks 1,2 for the scatter-adds) ---
  {
    long long n1 = (long long)ns * 48;
    k_init_s<<<(unsigned)((n1 + T - 1) / T), T, 0, stream>>>(ue, sie, s_all, NU, ns);
    long long n2 = (long long)nt * 16;
    k_init_t<<<(unsigned)((n2 + T - 1) / T), T, 0, stream>>>(ue, tie, tx_a, NU, nt);
  }
  hipMemsetAsync(tx_b, 0, (size_t)nt * DIM * sizeof(float), stream);

  unsigned convBlkS = (unsigned)(((long long)E * 16 + T - 1) / T);
  unsigned convBlkT = (unsigned)(((long long)Et * 16 + T - 1) / T);
  unsigned mixBlk   = (unsigned)((((NU + 15) / 16) * 32 + T - 1) / T);

  // --- layer 0 ---
  k_conv<<<convBlkS, T, 0, stream>>>(s_all,      s_all + DIM, s_src, s_dst, dinv_s, E,  SD,  SD);
  k_conv<<<convBlkT, T, 0, stream>>>(tx_a,       tx_b,        t_src, t_dst, dinv_t, Et, DIM, DIM);
  k_mix<<<mixBlk, T, 0, stream>>>(mix_W, mix_b, s_all + DIM, tx_b, NU);

  // --- layer 1 ---
  hipMemsetAsync(tx_a, 0, (size_t)nt * DIM * sizeof(float), stream);
  k_conv<<<convBlkS, T, 0, stream>>>(s_all + DIM, s_all + 2 * DIM, s_src, s_dst, dinv_s, E,  SD,  SD);
  k_conv<<<convBlkT, T, 0, stream>>>(tx_b,        tx_a,            t_src, t_dst, dinv_t, Et, DIM, DIM);
  k_mix<<<mixBlk, T, 0, stream>>>(mix_W + (size_t)DIM * 2 * DIM, mix_b + DIM, s_all + 2 * DIM, tx_a, NU);

  // --- prediction ---
  unsigned predBlk = (unsigned)(((long long)B * 32 + T - 1) / T);
  k_pred<<<predBlk, T, 0, stream>>>(s_all, l_u, l_i, pred_W, pred_b, (float*)d_out, B);
}
